// Electra_ETC_28501402976671
// MI455X (gfx1250) — matrix operand order repaired and checked by reading, compiled, not run on hardware
//
#include <hip/hip_runtime.h>
#include <cstddef>
#include <cstdint>

// ---------------------------------------------------------------------------
// Model dims (fixed by the reference)
// ---------------------------------------------------------------------------
#define BB   8
#define LL   1024
#define SS   1025
#define HH   256
#define FF_  1024
#define NLAY 12
#define NH_L 4      // local heads, dh = 64
#define NH_G 8      // global heads, dh = 32
#define BLH  (BB * LL * HH)       // 2,097,152
#define BLF  (BB * LL * FF_)      // 8,388,608

typedef __bf16 bf16;
typedef __attribute__((ext_vector_type(16))) __bf16 v16bf;
typedef __attribute__((ext_vector_type(8)))  __bf16 v8bf;
typedef __attribute__((ext_vector_type(8)))  float  v8f;

#define DEV_ static __device__ __forceinline__

DEV_ v8f wmma_bf16(v16bf a, v16bf b, v8f c) {
  // D = A(16x32 bf16) * B(32x16 bf16) + C(16x16 f32)
  return __builtin_amdgcn_wmma_f32_16x16x32_bf16(false, a, false, b, (short)0, c,
                                                 false, false);
}

// A-fragment: this lane supplies K = kb..kb+7 (elems 0..7) and kb+16..kb+23
// (elems 8..15) of its M-row, matching the 16-bit 16x32 A layout.
DEV_ v16bf load_a_frag(const bf16* p /* row base + kb */) {
  v8bf lo = *(const v8bf*)(p);
  v8bf hi = *(const v8bf*)(p + 16);
  return __builtin_shufflevector(lo, hi, 0, 1, 2, 3, 4, 5, 6, 7,
                                 8, 9, 10, 11, 12, 13, 14, 15);
}

DEV_ v16bf load_b_frag(const bf16* p, size_t stride) {
  v16bf b;
#pragma unroll
  for (int i = 0; i < 8; ++i) {
    b[i]     = p[(size_t)i * stride];
    b[8 + i] = p[(size_t)(16 + i) * stride];
  }
  return b;
}

DEV_ float gelu_exact(float x) {
  return 0.5f * x * (1.0f + erff(x * 0.70710678118654752f));
}

// ---------------------------------------------------------------------------
// f32 -> bf16 bulk convert
// ---------------------------------------------------------------------------
__global__ __launch_bounds__(256) void cvt_bf16_kernel(const float* __restrict__ s,
                                                       bf16* __restrict__ d, int n) {
  int i = blockIdx.x * 256 + threadIdx.x;
  if (i < n) d[i] = (bf16)s[i];
}

// ---------------------------------------------------------------------------
// Embedding gather + LN(E=128) + projection to H=256.  One wave per row.
// Row 0 of each sequence -> g,  rows 1.. -> loc (f32 + bf16).
// ---------------------------------------------------------------------------
__global__ __launch_bounds__(256) void embed_kernel(
    const int* __restrict__ xs, const float* __restrict__ emb_word,
    const float* __restrict__ emb_pos, const float* __restrict__ emb_tok,
    const float* __restrict__ lng, const float* __restrict__ lnb,
    const float* __restrict__ proj_w, const float* __restrict__ proj_b,
    float* __restrict__ g, float* __restrict__ loc, bf16* __restrict__ locbf) {
  int wave = threadIdx.x >> 5, lane = threadIdx.x & 31;
  int row = blockIdx.x * 8 + wave;
  if (row >= BB * SS) return;
  int b = row / SS, s = row % SS;
  int tok = xs[row];

  float e[4];
  float sum = 0.f;
#pragma unroll
  for (int i = 0; i < 4; ++i) {
    int k = lane + 32 * i;
    float t = emb_word[(size_t)tok * 128 + k] + emb_pos[(size_t)s * 128 + k] + emb_tok[k];
    e[i] = t;
    sum += t;
  }
#pragma unroll
  for (int o = 1; o < 32; o <<= 1) sum += __shfl_xor(sum, o, 32);
  float mean = sum * (1.0f / 128.0f);
  float var = 0.f;
#pragma unroll
  for (int i = 0; i < 4; ++i) { float d = e[i] - mean; var += d * d; }
#pragma unroll
  for (int o = 1; o < 32; o <<= 1) var += __shfl_xor(var, o, 32);
  float rstd = rsqrtf(var * (1.0f / 128.0f) + 1e-12f);
#pragma unroll
  for (int i = 0; i < 4; ++i) {
    int k = lane + 32 * i;
    e[i] = (e[i] - mean) * rstd * lng[k] + lnb[k];
  }

  float acc[8];
#pragma unroll
  for (int j = 0; j < 8; ++j) acc[j] = proj_b[lane + 32 * j];
#pragma unroll
  for (int r = 0; r < 4; ++r)
    for (int k2 = 0; k2 < 32; ++k2) {
      float ek = __shfl(e[r], k2, 32);
      const float* w = proj_w + (size_t)(r * 32 + k2) * HH + lane;
#pragma unroll
      for (int j = 0; j < 8; ++j) acc[j] += ek * w[32 * j];
    }

  if (s == 0) {
#pragma unroll
    for (int j = 0; j < 8; ++j) g[b * HH + lane + 32 * j] = acc[j];
  } else {
    size_t o = ((size_t)b * LL + (s - 1)) * HH;
#pragma unroll
    for (int j = 0; j < 8; ++j) {
      loc[o + lane + 32 * j] = acc[j];
      locbf[o + lane + 32 * j] = (bf16)acc[j];
    }
  }
}

// lm[b,i] = valid ? 0 : -1e9  (validity from xs body tokens)
__global__ __launch_bounds__(256) void lm_kernel(const int* __restrict__ xs,
                                                 float* __restrict__ lm) {
  int j = blockIdx.x * 256 + threadIdx.x;  // 8192
  int b = j >> 10, i = j & 1023;
  lm[j] = (xs[b * SS + 1 + i] > 0) ? 0.0f : -1.0e9f;
}

// ---------------------------------------------------------------------------
// Generic bf16 WMMA GEMM, templated on K,N so all addressing is static.
// 8 waves/block; each wave owns a 32x64 strip (2 A-frags x 4 B-frags ->
// 8 wmma per 32-wide K step).  All loads for a step issue before the wmmas.
// Epilogue modes select output format.
// ---------------------------------------------------------------------------
#define GM_F32  0
#define GM_BF16 1
#define GM_GELU 2
#define GM_KT   3   // bf16 write transposed to kt[(b*4+h)*64+d][key]

template <int K, int N>
__global__ __launch_bounds__(256) void gemm_bf16_kernel(
    const bf16* __restrict__ A, const bf16* __restrict__ W,
    const float* __restrict__ bias, void* __restrict__ Cout, int mode) {
  int tid = threadIdx.x;
  int wave = tid >> 5, lane = tid & 31;
  int hf = lane >> 4, r16 = lane & 15;
  int m0 = blockIdx.x * 256 + wave * 32;
  int n0 = blockIdx.y * 64;

  v8f acc[8] = {};
  const bf16* Arow0 = A + (size_t)(m0 + r16) * K + hf * 8;
  const bf16* Arow1 = Arow0 + 16 * K;
  const bf16* Brow = W + (size_t)(hf * 8) * N + n0 + r16;

#pragma unroll 2
  for (int k = 0; k < K; k += 32) {
    v16bf a0 = load_a_frag(Arow0 + k);
    v16bf a1 = load_a_frag(Arow1 + k);
    const bf16* Bp = Brow + (size_t)k * N;
    v16bf b0 = load_b_frag(Bp, N);
    v16bf b1 = load_b_frag(Bp + 16, N);
    v16bf b2 = load_b_frag(Bp + 32, N);
    v16bf b3 = load_b_frag(Bp + 48, N);
    acc[0] = wmma_bf16(a0, b0, acc[0]);
    acc[1] = wmma_bf16(a0, b1, acc[1]);
    acc[2] = wmma_bf16(a0, b2, acc[2]);
    acc[3] = wmma_bf16(a0, b3, acc[3]);
    acc[4] = wmma_bf16(a1, b0, acc[4]);
    acc[5] = wmma_bf16(a1, b1, acc[5]);
    acc[6] = wmma_bf16(a1, b2, acc[6]);
    acc[7] = wmma_bf16(a1, b3, acc[7]);
  }

#pragma unroll
  for (int gblk = 0; gblk < 2; ++gblk) {
#pragma unroll
    for (int t = 0; t < 4; ++t) {
      int n = n0 + t * 16 + r16;
      float bv = bias ? bias[n] : 0.0f;
      v8f& a = acc[gblk * 4 + t];
#pragma unroll
      for (int r = 0; r < 8; ++r) {
        int m = m0 + gblk * 16 + r + hf * 8;
        float v = a[r] + bv;
        if (mode == GM_F32) {
          ((float*)Cout)[(size_t)m * N + n] = v;
        } else if (mode == GM_BF16) {
          ((bf16*)Cout)[(size_t)m * N + n] = (bf16)v;
        } else if (mode == GM_GELU) {
          ((bf16*)Cout)[(size_t)m * N + n] = (bf16)gelu_exact(v);
        } else {  // GM_KT : m = b*L+key, n = h*64+d
          int b = m >> 10, key = m & 1023;
          int h = n >> 6, d = n & 63;
          ((bf16*)Cout)[(((size_t)b * NH_L + h) * 64 + d) * LL + key] = (bf16)v;
        }
      }
    }
  }
}

// ---------------------------------------------------------------------------
// Local attention: grid (qb=64, h=4, b=8); block 256 (8 waves).
// Phase1 QK^T (WMMA) -> 16x1024 f32 LDS ; Phase2 masked softmax in LDS ;
// Phase3 P*V (WMMA, P converted f32->bf16 from LDS) -> ctx bf16.
// ---------------------------------------------------------------------------
__global__ __launch_bounds__(256) void local_attn_kernel(
    const bf16* __restrict__ q, const bf16* __restrict__ kt,
    const bf16* __restrict__ v, const float* __restrict__ lm,
    bf16* __restrict__ ctx) {
  __shared__ float sc[16 * 1024];  // 64KB
  int b = blockIdx.z, h = blockIdx.y, qb = blockIdx.x;
  int tid = threadIdx.x;
  int wave = tid >> 5, lane = tid & 31;
  int hf = lane >> 4, r16 = lane & 15;
  const float scale = 0.125f;  // 1/sqrt(64)

  // ---- Phase 1: scores ----
  const bf16* qrow = q + ((size_t)(b * LL + qb * 16 + r16)) * HH + h * 64;
  v16bf aq[2];
#pragma unroll
  for (int kk = 0; kk < 2; ++kk) aq[kk] = load_a_frag(qrow + kk * 32 + hf * 8);

  const bf16* ktbase = kt + (((size_t)b * NH_L + h) * 64) * LL;
  for (int ti = wave * 8; ti < wave * 8 + 8; ++ti) {
    int key0 = ti * 16;
    v16bf bf0 = load_b_frag(ktbase + (size_t)(hf * 8) * LL + key0 + r16, LL);
    v16bf bf1 = load_b_frag(ktbase + (size_t)(32 + hf * 8) * LL + key0 + r16, LL);
    v8f c = {};
    c = wmma_bf16(aq[0], bf0, c);
    c = wmma_bf16(aq[1], bf1, c);
    int key = key0 + r16;
    float lmv = lm[b * LL + key];
#pragma unroll
    for (int r = 0; r < 8; ++r) sc[(r + hf * 8) * 1024 + key] = c[r] * scale + lmv;
  }
  __syncthreads();

  // ---- Phase 2: softmax (16 threads per row, shuffle reduce within 16) ----
  {
    int row = tid >> 4;
    int c0 = tid & 15;
    float m = -1.0e30f;
    for (int j = c0; j < 1024; j += 16) m = fmaxf(m, sc[row * 1024 + j]);
#pragma unroll
    for (int o = 1; o < 16; o <<= 1) m = fmaxf(m, __shfl_xor(m, o, 32));
    float sum = 0.f;
    for (int j = c0; j < 1024; j += 16) {
      float p = __expf(sc[row * 1024 + j] - m);
      sc[row * 1024 + j] = p;
      sum += p;
    }
#pragma unroll
    for (int o = 1; o < 16; o <<= 1) sum += __shfl_xor(sum, o, 32);
    float inv = 1.0f / sum;
    for (int j = c0; j < 1024; j += 16) sc[row * 1024 + j] *= inv;
  }
  __syncthreads();

  // ---- Phase 3: ctx = P * V  (waves 0..3, one 16x16 d-tile each) ----
  if (wave < 4) {
    int n0 = wave * 16;
    v8f c = {};
    for (int k = 0; k < 1024; k += 32) {
      v16bf a;
      int kb = k + hf * 8;
#pragma unroll
      for (int i = 0; i < 8; ++i) {
        a[i]     = (bf16)sc[r16 * 1024 + kb + i];
        a[8 + i] = (bf16)sc[r16 * 1024 + kb + 16 + i];
      }
      v16bf bf = load_b_frag(v + ((size_t)(b * LL + k + hf * 8)) * HH + h * 64 + n0 + r16, HH);
      c = wmma_bf16(a, bf, c);
    }
#pragma unroll
    for (int r = 0; r < 8; ++r) {
      int row = r + hf * 8;
      ctx[((size_t)(b * LL + qb * 16 + row)) * HH + h * 64 + n0 + r16] = (bf16)c[r];
    }
  }
}

// ---------------------------------------------------------------------------
// LayerNorm over H=256 for 8192 rows: out = LN(x + res); writes f32/bf16.
// ---------------------------------------------------------------------------
__global__ __launch_bounds__(256) void ln_big_kernel(
    const float* __restrict__ x, const float* __restrict__ res,
    const float* __restrict__ gamma, const float* __restrict__ beta, float eps,
    float* __restrict__ out32, bf16* __restrict__ outbf) {
  int wave = threadIdx.x >> 5, lane = threadIdx.x & 31;
  size_t row = (size_t)blockIdx.x * 8 + wave;
  const float* xr = x + row * HH;
  const float* rr = res + row * HH;
  float vals[8];
  float s = 0.f;
#pragma unroll
  for (int i = 0; i < 8; ++i) {
    float t = xr[lane + 32 * i] + rr[lane + 32 * i];
    vals[i] = t;
    s += t;
  }
#pragma unroll
  for (int o = 1; o < 32; o <<= 1) s += __shfl_xor(s, o, 32);
  float mean = s * (1.0f / HH);
  float var = 0.f;
#pragma unroll
  for (int i = 0; i < 8; ++i) { float d = vals[i] - mean; var += d * d; }
#pragma unroll
  for (int o = 1; o < 32; o <<= 1) var += __shfl_xor(var, o, 32);
  float rstd = rsqrtf(var * (1.0f / HH) + eps);
#pragma unroll
  for (int i = 0; i < 8; ++i) {
    int k = lane + 32 * i;
    float o = (vals[i] - mean) * rstd * gamma[k] + beta[k];
    if (out32) out32[row * HH + k] = o;
    if (outbf) outbf[row * HH + k] = (bf16)o;
  }
}

// Small LN: 8 rows x 256, one block, wave per row.
__global__ __launch_bounds__(256) void ln_small_kernel(
    const float* __restrict__ x, const float* __restrict__ res,
    const float* __restrict__ gamma, const float* __restrict__ beta, float eps,
    float* __restrict__ out) {
  int row = threadIdx.x >> 5, lane = threadIdx.x & 31;
  const float* xr = x + row * HH;
  const float* rr = res ? res + row * HH : nullptr;
  float vals[8];
  float s = 0.f;
#pragma unroll
  for (int i = 0; i < 8; ++i) {
    float t = xr[lane + 32 * i] + (rr ? rr[lane + 32 * i] : 0.f);
    vals[i] = t;
    s += t;
  }
#pragma unroll
  for (int o = 1; o < 32; o <<= 1) s += __shfl_xor(s, o, 32);
  float mean = s * (1.0f / HH);
  float var = 0.f;
#pragma unroll
  for (int i = 0; i < 8; ++i) { float d = vals[i] - mean; var += d * d; }
#pragma unroll
  for (int o = 1; o < 32; o <<= 1) var += __shfl_xor(var, o, 32);
  float rstd = rsqrtf(var * (1.0f / HH) + eps);
#pragma unroll
  for (int i = 0; i < 8; ++i) {
    int k = lane + 32 * i;
    out[row * HH + k] = (vals[i] - mean) * rstd * gamma[k] + beta[k];
  }
}

// Small f32 linear for the 8-row global path: C = act(A@W + b)
__global__ __launch_bounds__(256) void small_linear_kernel(
    const float* __restrict__ A, const float* __restrict__ W,
    const float* __restrict__ bias, float* __restrict__ C,
    int M, int N, int K, int act) {
  int idx = blockIdx.x * 256 + threadIdx.x;
  if (idx >= M * N) return;
  int m = idx / N, n = idx % N;
  float acc = bias ? bias[n] : 0.0f;
  for (int k = 0; k < K; ++k) acc += A[(size_t)m * K + k] * W[(size_t)k * N + n];
  if (act == 1) acc = gelu_exact(acc);
  else if (act == 2) acc = fmaxf(acc, 0.0f);
  C[idx] = acc;
}

// ---------------------------------------------------------------------------
// g->local attention: one block per (b,h); 8192 keys, two-pass softmax in LDS
// ---------------------------------------------------------------------------
__global__ __launch_bounds__(256) void g2l_attn_kernel(
    const float* __restrict__ qg, const bf16* __restrict__ kg,
    const bf16* __restrict__ vg, const float* __restrict__ lm,
    float* __restrict__ cg) {
  __shared__ float sc[BB * LL];
  __shared__ float qs[64];
  __shared__ float red[256];
  int b = blockIdx.x, h = blockIdx.y;
  int tid = threadIdx.x;
  if (tid < 64) qs[tid] = qg[b * HH + h * 64 + tid];
  __syncthreads();

  float m = -1.0e30f;
  for (int j = tid; j < BB * LL; j += 256) {
    const bf16* kr = kg + (size_t)j * HH + h * 64;
    float dot = 0.f;
#pragma unroll 8
    for (int d = 0; d < 64; ++d) dot += qs[d] * (float)kr[d];
    int jb = j >> 10;
    float mask = (jb <= b && lm[j] == 0.0f) ? 0.0f : -10000.0f;
    float s = dot * 0.125f + mask;
    sc[j] = s;
    m = fmaxf(m, s);
  }
  red[tid] = m;
  __syncthreads();
  for (int s = 128; s > 0; s >>= 1) {
    if (tid < s) red[tid] = fmaxf(red[tid], red[tid + s]);
    __syncthreads();
  }
  m = red[0];
  __syncthreads();

  float sum = 0.f;
  for (int j = tid; j < BB * LL; j += 256) {
    float p = __expf(sc[j] - m);
    sc[j] = p;
    sum += p;
  }
  red[tid] = sum;
  __syncthreads();
  for (int s = 128; s > 0; s >>= 1) {
    if (tid < s) red[tid] += red[tid + s];
    __syncthreads();
  }
  float inv = 1.0f / red[0];
  __syncthreads();

  int d = tid & 63, grp = tid >> 6;
  float acc = 0.f;
  for (int j = grp; j < BB * LL; j += 4)
    acc += sc[j] * (float)vg[(size_t)j * HH + h * 64 + d];
  red[tid] = acc;
  __syncthreads();
  if (grp == 0) {
    float t = red[d] + red[64 + d] + red[128 + d] + red[192 + d];
    cg[b * HH + h * 64 + d] = t * inv;
  }
}

// 8x8 global attention (NH_G=8 heads of 32), trilm is an ADDITIVE tril(ones).
__global__ __launch_bounds__(256) void global_attn_kernel(
    const float* __restrict__ qkv, float* __restrict__ c2) {
  __shared__ float p[8][8][8];  // [b][h][c]
  int tid = threadIdx.x;
  if (tid < 64) {
    int b = tid >> 3, h = tid & 7;
    const float* qv = qkv + b * 768 + h * 32;
    float s[8];
    float m = -1.0e30f;
#pragma unroll
    for (int c = 0; c < 8; ++c) {
      const float* kv = qkv + c * 768 + 256 + h * 32;
      float dot = 0.f;
#pragma unroll
      for (int d = 0; d < 32; ++d) dot += qv[d] * kv[d];
      s[c] = dot * 0.17677669529663687f + ((c <= b) ? 1.0f : 0.0f);
      m = fmaxf(m, s[c]);
    }
    float sum = 0.f;
#pragma unroll
    for (int c = 0; c < 8; ++c) { s[c] = __expf(s[c] - m); sum += s[c]; }
    float inv = 1.0f / sum;
#pragma unroll
    for (int c = 0; c < 8; ++c) p[b][h][c] = s[c] * inv;
  }
  __syncthreads();
  int col = tid, h = col >> 5, d = col & 31;
  for (int b = 0; b < 8; ++b) {
    float acc = 0.f;
#pragma unroll
    for (int c = 0; c < 8; ++c) acc += p[b][h][c] * qkv[c * 768 + 512 + h * 32 + d];
    c2[b * HH + col] = acc;
  }
}

// loc = broadcast(new_loc), f32 + bf16
__global__ __launch_bounds__(256) void broadcast_loc_kernel(
    const float* __restrict__ nl, float* __restrict__ loc, bf16* __restrict__ locbf) {
  size_t idx = (size_t)blockIdx.x * 256 + threadIdx.x;  // BLH
  int b = (int)(idx >> 18);  // L*H = 2^18
  int c = (int)(idx & 255);
  float v = nl[b * HH + c];
  loc[idx] = v;
  locbf[idx] = (bf16)v;
}

// g += pe  (sinusoidal over batch index)
__global__ __launch_bounds__(256) void add_pe_kernel(float* __restrict__ g) {
  int b = blockIdx.x, d = threadIdx.x;
  int i = d >> 1;
  float div = __expf((float)(2 * i) * (-9.210340371976184f / 256.0f));
  float ang = (float)b * div;
  g[b * HH + d] += (d & 1) ? cosf(ang) : sinf(ang);
}

// ---------------------------------------------------------------------------
// Host side
// ---------------------------------------------------------------------------
extern "C" void kernel_launch(void* const* d_in, const int* in_sizes, int n_in,
                              void* d_out, int out_size, void* d_ws, size_t ws_size,
                              hipStream_t stream) {
  (void)in_sizes; (void)n_in; (void)out_size; (void)ws_size;
  const int*   xs       = (const int*)d_in[0];
  const float* emb_word = (const float*)d_in[1];
  const float* emb_pos  = (const float*)d_in[2];
  const float* emb_tok  = (const float*)d_in[3];
  const float* emb_ln_g = (const float*)d_in[4];
  const float* emb_ln_b = (const float*)d_in[5];
  const float* proj_w   = (const float*)d_in[6];
  const float* proj_b   = (const float*)d_in[7];
  const float* Lq_w = (const float*)d_in[8];  const float* Lq_b = (const float*)d_in[9];
  const float* Lk_w = (const float*)d_in[10]; const float* Lk_b = (const float*)d_in[11];
  const float* Lv_w = (const float*)d_in[12]; const float* Lv_b = (const float*)d_in[13];
  const float* Lo_w = (const float*)d_in[14]; const float* Lo_b = (const float*)d_in[15];
  const float* Lln1_g = (const float*)d_in[16]; const float* Lln1_b = (const float*)d_in[17];
  const float* Li_w = (const float*)d_in[18]; const float* Li_b = (const float*)d_in[19];
  const float* Lo2_w = (const float*)d_in[20]; const float* Lo2_b = (const float*)d_in[21];
  const float* Lln2_g = (const float*)d_in[22]; const float* Lln2_b = (const float*)d_in[23];
  const float* Gqkv_w = (const float*)d_in[24]; const float* Gqkv_b = (const float*)d_in[25];
  const float* Gout_w = (const float*)d_in[26]; const float* Gout_b = (const float*)d_in[27];
  const float* Gln1_g = (const float*)d_in[28]; const float* Gln1_b = (const float*)d_in[29];
  const float* Gln2_g = (const float*)d_in[30]; const float* Gln2_b = (const float*)d_in[31];
  const float* G1_w = (const float*)d_in[32]; const float* G1_b = (const float*)d_in[33];
  const float* G2_w = (const float*)d_in[34]; const float* G2_b = (const float*)d_in[35];
  const float* cls_w = (const float*)d_in[36]; const float* cls_b = (const float*)d_in[37];

  // ---- bump allocator over d_ws ----
  char* ws = (char*)d_ws;
  size_t off = 0;
  auto alloc = [&](size_t bytes) -> void* {
    void* p = ws + off;
    off = (off + bytes + 255) & ~(size_t)255;
    return p;
  };
  float* g_buf  = (float*)alloc(BB * HH * 4);
  float* loc32  = (float*)alloc((size_t)BLH * 4);
  bf16*  locbf  = (bf16*)alloc((size_t)BLH * 2);
  float* lm     = (float*)alloc(BB * LL * 4);
  bf16*  LqWbf  = (bf16*)alloc((size_t)NLAY * HH * HH * 2);
  bf16*  LkWbf  = (bf16*)alloc((size_t)NLAY * HH * HH * 2);
  bf16*  LvWbf  = (bf16*)alloc((size_t)NLAY * HH * HH * 2);
  bf16*  LoWbf  = (bf16*)alloc((size_t)NLAY * HH * HH * 2);
  bf16*  LiWbf  = (bf16*)alloc((size_t)NLAY * HH * FF_ * 2);
  bf16*  Lo2Wbf = (bf16*)alloc((size_t)NLAY * FF_ * HH * 2);
  bf16*  bufA   = (bf16*)alloc((size_t)BLH * 2);
  bf16*  bufB   = (bf16*)alloc((size_t)BLH * 2);
  bf16*  bufC   = (bf16*)alloc((size_t)BLH * 2);
  bf16*  ktbf   = (bf16*)alloc((size_t)BLH * 2);
  bf16*  ffbf   = (bf16*)alloc((size_t)BLF * 2);
  float* f32buf = (float*)alloc((size_t)BLH * 4);
  float* a32buf = (float*)alloc((size_t)BLH * 4);
  float* s_t    = (float*)alloc(BB * HH * 4);
  float* s_a2   = (float*)alloc(BB * HH * 4);
  float* s_ff   = (float*)alloc(BB * FF_ * 4);
  float* s_nl   = (float*)alloc(BB * HH * 4);
  float* s_qg   = (float*)alloc(BB * HH * 4);
  float* s_cg   = (float*)alloc(BB * HH * 4);
  float* s_a3   = (float*)alloc(BB * HH * 4);
  float* s_qkv  = (float*)alloc(BB * 768 * 4);
  float* s_c2   = (float*)alloc(BB * HH * 4);
  float* s_x1   = (float*)alloc(BB * HH * 4);

  auto cvt = [&](const float* s, bf16* d, int n) {
    cvt_bf16_kernel<<<(n + 255) / 256, 256, 0, stream>>>(s, d, n);
  };
  const int M = BB * LL;  // 8192
  // Templated GEMM dispatch: (K,N) in {(256,256), (256,1024), (1024,256)}
  auto gemm_h_h = [&](const bf16* A, const bf16* W, const float* bias, void* C, int mode) {
    gemm_bf16_kernel<HH, HH><<<dim3(M / 256, HH / 64), 256, 0, stream>>>(A, W, bias, C, mode);
  };
  auto gemm_h_f = [&](const bf16* A, const bf16* W, const float* bias, void* C, int mode) {
    gemm_bf16_kernel<HH, FF_><<<dim3(M / 256, FF_ / 64), 256, 0, stream>>>(A, W, bias, C, mode);
  };
  auto gemm_f_h = [&](const bf16* A, const bf16* W, const float* bias, void* C, int mode) {
    gemm_bf16_kernel<FF_, HH><<<dim3(M / 256, HH / 64), 256, 0, stream>>>(A, W, bias, C, mode);
  };
  auto slin = [&](const float* A, const float* W, const float* bias, float* C,
                  int Ms, int Ns, int Ks, int act) {
    small_linear_kernel<<<(Ms * Ns + 255) / 256, 256, 0, stream>>>(A, W, bias, C, Ms, Ns, Ks, act);
  };
  auto lnS = [&](const float* x, const float* res, const float* ga, const float* be,
                 float eps, float* out) {
    ln_small_kernel<<<1, 256, 0, stream>>>(x, res, ga, be, eps, out);
  };

  // ---- setup ----
  cvt(Lq_w, LqWbf, NLAY * HH * HH);
  cvt(Lk_w, LkWbf, NLAY * HH * HH);
  cvt(Lv_w, LvWbf, NLAY * HH * HH);
  cvt(Lo_w, LoWbf, NLAY * HH * HH);
  cvt(Li_w, LiWbf, NLAY * HH * FF_);
  cvt(Lo2_w, Lo2Wbf, NLAY * FF_ * HH);
  embed_kernel<<<(BB * SS + 7) / 8, 256, 0, stream>>>(
      xs, emb_word, emb_pos, emb_tok, emb_ln_g, emb_ln_b, proj_w, proj_b,
      g_buf, loc32, locbf);
  lm_kernel<<<BB * LL / 256, 256, 0, stream>>>(xs, lm);

  for (int l = 0; l < NLAY; ++l) {
    const size_t wHH = (size_t)l * HH * HH, wHF = (size_t)l * HH * FF_;
    const float *lqb = Lq_b + l * HH, *lkb = Lk_b + l * HH, *lvb = Lv_b + l * HH,
                *lob = Lo_b + l * HH, *lib = Li_b + l * FF_, *lo2b = Lo2_b + l * HH;
    const float *ln1g = Lln1_g + l * HH, *ln1b = Lln1_b + l * HH,
                *ln2g = Lln2_g + l * HH, *ln2b = Lln2_b + l * HH;

    // ---- local token path (big, WMMA) ----
    gemm_h_h(locbf, LqWbf + wHH, lqb, bufA, GM_BF16);                     // q
    gemm_h_h(locbf, LkWbf + wHH, lkb, ktbf, GM_KT);                       // k^T
    gemm_h_h(locbf, LvWbf + wHH, lvb, bufB, GM_BF16);                     // v
    local_attn_kernel<<<dim3(LL / 16, NH_L, BB), 256, 0, stream>>>(
        bufA, ktbf, bufB, lm, bufC);                                      // ctx
    gemm_h_h(bufC, LoWbf + wHH, lob, f32buf, GM_F32);                     // ctx@Lo
    ln_big_kernel<<<M / 8, 256, 0, stream>>>(f32buf, loc32, ln1g, ln1b,
                                             1e-12f, a32buf, bufA);       // a
    gemm_h_f(bufA, LiWbf + wHF, lib, ffbf, GM_GELU);                      // gelu(a@Li)
    gemm_f_h(ffbf, Lo2Wbf + wHF, lo2b, f32buf, GM_F32);                   // @Lo2
    ln_big_kernel<<<M / 8, 256, 0, stream>>>(f32buf, a32buf, ln2g, ln2b,
                                             1e-12f, nullptr, bufB);      // Lp
    gemm_h_h(bufB, LkWbf + wHH, lkb, bufC, GM_BF16);                      // kg
    gemm_h_h(bufB, LvWbf + wHH, lvb, bufA, GM_BF16);                      // vg

    // ---- global token path (small, VALU) ----
    slin(g_buf, Lv_w + wHH, lvb, s_t, BB, HH, HH, 0);                     // ctx_g
    slin(s_t, Lo_w + wHH, lob, s_a2, BB, HH, HH, 0);
    lnS(s_a2, g_buf, ln1g, ln1b, 1e-12f, s_a2);                           // a2
    slin(s_a2, Li_w + wHF, lib, s_ff, BB, FF_, HH, 1);
    slin(s_ff, Lo2_w + wHF, lo2b, s_t, BB, HH, FF_, 0);
    lnS(s_t, s_a2, ln2g, ln2b, 1e-12f, s_nl);                             // new_loc
    slin(g_buf, Lq_w + wHH, lqb, s_qg, BB, HH, HH, 0);                    // qg
    g2l_attn_kernel<<<dim3(BB, NH_L), 256, 0, stream>>>(s_qg, bufC, bufA, lm, s_cg);
    slin(s_cg, Lo_w + wHH, lob, s_t, BB, HH, HH, 0);
    lnS(s_t, g_buf, ln1g, ln1b, 1e-12f, s_a3);                            // a3
    slin(s_a3, Li_w + wHF, lib, s_ff, BB, FF_, HH, 1);
    slin(s_ff, Lo2_w + wHF, lo2b, s_t, BB, HH, FF_, 0);
    lnS(s_t, s_a3, ln2g, ln2b, 1e-12f, g_buf);                            // g = ffn(a3)
    broadcast_loc_kernel<<<BLH / 256, 256, 0, stream>>>(s_nl, loc32, locbf);
    if (l == 0) add_pe_kernel<<<BB, HH, 0, stream>>>(g_buf);

    // ---- global self-attention over the 8 tokens ----
    slin(g_buf, Gqkv_w + (size_t)l * HH * 768, Gqkv_b + l * 768, s_qkv, BB, 768, HH, 0);
    global_attn_kernel<<<1, 256, 0, stream>>>(s_qkv, s_c2);
    slin(s_c2, Gout_w + wHH, Gout_b + l * HH, s_t, BB, HH, HH, 0);
    lnS(s_t, g_buf, Gln1_g + l * HH, Gln1_b + l * HH, 1e-5f, s_x1);       // x1
    slin(s_x1, G1_w + wHF, G1_b + l * FF_, s_ff, BB, FF_, HH, 2);         // relu
    slin(s_ff, G2_w + wHF, G2_b + l * HH, s_t, BB, HH, FF_, 0);
    lnS(s_t, s_x1, Gln2_g + l * HH, Gln2_b + l * HH, 1e-5f, g_buf);       // g
  }

  // classifier -> d_out (8 x 1 x 2, f32)
  slin(g_buf, cls_w, cls_b, (float*)d_out, BB, 2, HH, 0);
}
